// TransformerBlock_59304908423836
// MI455X (gfx1250) — matrix operand order repaired and checked by reading, not compiled, not run on hardware
//
#include <hip/hip_runtime.h>

// ---------------------------------------------------------------------------
// Transformer block forward for MI455X (gfx1250, wave32, WMMA).
// All matmuls run on v_wmma_f32_16x16x32_bf16 (fp32 accumulate).
// Residuals / LayerNorm / softmax statistics stay fp32.
// Working set (<230MB) is L2-resident (192MB L2, 23.3TB/s HBM).
// GEMM B-tiles are staged to LDS with GLOBAL_LOAD_ASYNC_TO_LDS_B128
// (ASYNCcnt ping-pong) so both wave-rows of a block share one L2 read;
// A fragments stream from L2 with contiguous-k 16B loads + global_prefetch.
// Attention: block-wide softmax stabilizer (valid for every row; softmax is
// invariant to the stabilizer) -> 5-shuffle full-wave max reduction instead
// of 32 serialized per-row ds_bpermutes; row sums via a ones-matrix WMMA.
// ---------------------------------------------------------------------------

typedef __bf16 bf16_t;
typedef __attribute__((ext_vector_type(16))) __bf16 v16bf;
typedef __attribute__((ext_vector_type(8)))  __bf16 v8bf;
typedef __attribute__((ext_vector_type(8)))  float  v8f;

#define NB    4
#define SEQ   2048
#define DM    1024
#define NH    16
#define HDIM  64
#define DFF   3072
#define MROWS (NB * SEQ)   // 8192

union FragU { v16bf v; v8bf h[2]; };

// A-fragment (16x32 bf16, MxK): lane l holds row m=l&15.
// half = l>>4: elements 0..15 <- K = half*16 + 0..15 (contiguous, matches B)
__device__ __forceinline__ v16bf load_a_frag(const bf16_t* p, int half) {
  FragU u;
  u.h[0] = *reinterpret_cast<const v8bf*>(p + half * 16);
  u.h[1] = *reinterpret_cast<const v8bf*>(p + half * 16 + 8);
  return u.v;
}

// B-fragment (32x16 bf16, KxN): lane l holds column n=l&15.
// half = l>>4: elements 0..15 <- K = half*16 + 0..15 (contiguous).
// p points at row n of Bt (= B^T row-major) at the current K block.
__device__ __forceinline__ v16bf load_b_frag(const bf16_t* p, int half) {
  FragU u;
  u.h[0] = *reinterpret_cast<const v8bf*>(p + half * 16);
  u.h[1] = *reinterpret_cast<const v8bf*>(p + half * 16 + 8);
  return u.v;
}

__device__ __forceinline__ v8f wmma_bf(v16bf a, v16bf b, v8f c) {
  return __builtin_amdgcn_wmma_f32_16x16x32_bf16(
      false, a, false, b, (short)0, c, false, false);
}

__device__ __forceinline__ v16bf ones_frag() {
  FragU u;
  const bf16_t one = (bf16_t)1.0f;
#pragma unroll
  for (int i = 0; i < 8; ++i) { u.h[0][i] = one; u.h[1][i] = one; }
  return u.v;
}

// Async copy 16B global -> LDS (tracked by ASYNCcnt).
// lds_off: LDS byte address (low 32 bits of generic shared pointer).
__device__ __forceinline__ void async_b128(unsigned lds_off, const void* g) {
  asm volatile("global_load_async_to_lds_b128 %0, %1, off"
               :: "v"(lds_off), "v"((unsigned long long)(uintptr_t)g)
               : "memory");
}

// ---------------------------------------------------------------------------
// Weight convert + transpose: Wt[n][k] = (bf16) W[k][n].  Grid (K/32, N/32).
// ---------------------------------------------------------------------------
__global__ __launch_bounds__(256) void wconv_kernel(
    const float* __restrict__ W, bf16_t* __restrict__ Wt, int K, int N)
{
  __shared__ float tile[32][33];
  const int k0 = blockIdx.x * 32, n0 = blockIdx.y * 32;
  const int tx = threadIdx.x & 31, ty = threadIdx.x >> 5;
#pragma unroll
  for (int j = 0; j < 4; ++j)
    tile[ty + j * 8][tx] = W[(size_t)(k0 + ty + j * 8) * N + n0 + tx];
  __syncthreads();
#pragma unroll
  for (int j = 0; j < 4; ++j)
    Wt[(size_t)(n0 + ty + j * 8) * K + k0 + tx] = (bf16_t)tile[tx][ty + j * 8];
}

// ---------------------------------------------------------------------------
// Fused (optional pos-embed) + LayerNorm. One row (D=1024) per 256-thr block.
// ---------------------------------------------------------------------------
template <bool PE>
__global__ __launch_bounds__(256) void ln_kernel(
    const float* __restrict__ xin,
    const float* __restrict__ g, const float* __restrict__ be,
    float* __restrict__ xsave, bf16_t* __restrict__ outH)
{
  const int row = blockIdx.x;
  const int s   = row & (SEQ - 1);
  const float* xr = xin + (size_t)row * DM;
  float vals[4];
  float sum = 0.f, sq = 0.f;
#pragma unroll
  for (int j = 0; j < 4; ++j) {
    const int i = j * 256 + threadIdx.x;
    float v = xr[i];
    if (PE) {
      // 10000^(-2i/D) = exp(-(2i/D) * ln 1e4)
      float freq = __expf((float)i * (-2.0f * 9.210340371976184f / (float)DM));
      float ang  = (float)s * freq;
      v += (i & 1) ? __cosf(ang) : __sinf(ang);
      xsave[(size_t)row * DM + i] = v;
    }
    vals[j] = v;
    sum += v; sq += v * v;
  }
  __shared__ float red[16];
  const int lane = threadIdx.x & 31, wave = threadIdx.x >> 5;
#pragma unroll
  for (int off = 16; off >= 1; off >>= 1) {
    sum += __shfl_xor(sum, off);
    sq  += __shfl_xor(sq,  off);
  }
  if (lane == 0) { red[wave] = sum; red[8 + wave] = sq; }
  __syncthreads();
  float ts = 0.f, tq = 0.f;
#pragma unroll
  for (int w = 0; w < 8; ++w) { ts += red[w]; tq += red[8 + w]; }
  const float mu   = ts * (1.0f / DM);
  const float var  = tq * (1.0f / DM) - mu * mu;
  const float rstd = rsqrtf(var + 1e-5f);
#pragma unroll
  for (int j = 0; j < 4; ++j) {
    const int i = j * 256 + threadIdx.x;
    outH[(size_t)row * DM + i] = (bf16_t)((vals[j] - mu) * rstd * g[i] + be[i]);
  }
}

// ---------------------------------------------------------------------------
// bf16 WMMA GEMM: C(MxN) = A(MxK) * B(KxN), B given transposed (Bt = NxK).
// 256 threads = 8 waves; block tile 128x128; wave tile 64x32 (4x2 wmma accs).
// The 128x32 B slice is async-DMA'd to LDS (2-deep ping-pong, ASYNCcnt).
// MODE 0: out bf16 row-major          (Q,K projections)
// MODE 1: out bf16 transposed (NxM)   (V projection, for attention B-frags)
// MODE 2: outF = resid + acc          (attention output proj + residual)
// MODE 3: out bf16 = leakyrelu(acc+b) (FFN layer 1)
// MODE 4: outF = resid + acc + b      (FFN layer 2, final output)
// ---------------------------------------------------------------------------
template <int MODE>
__global__ __launch_bounds__(256) void gemm_kernel(
    const bf16_t* __restrict__ A,  int lda,
    const bf16_t* __restrict__ Bt, int ldb,
    const float* __restrict__ bias,
    const float* __restrict__ resid,
    float* __restrict__ outF, bf16_t* __restrict__ outH,
    int M, int N, int K)
{
  const int t    = threadIdx.x;
  const int lane = t & 31;
  const int wave = t >> 5;
  const int half = lane >> 4;
  const int ln   = lane & 15;
  const int m0 = blockIdx.y * 128 + (wave >> 2) * 64;
  const int n0w = (wave & 3) * 32;                 // wave's n offset in block
  const int n0 = blockIdx.x * 128 + n0w;

  // LDS staging for B: 2 buffers x (128 rows x 32 k) bf16 = 2 x 8KB.
  __shared__ __align__(16) bf16_t lbs[2][128 * 32];

  // Each thread DMAs one 32B slice per stage: row t>>1, k-half (t&1)*16.
  const int crow  = t >> 1;
  const int chalf = (t & 1) * 16;
  const bf16_t* gB = Bt + (size_t)(blockIdx.x * 128 + crow) * ldb + chalf;
  const unsigned loff =
      (unsigned)(uintptr_t)(&lbs[0][crow * 32 + chalf]);

  // prologue: stage 0 into buffer 0
  async_b128(loff,      gB);
  async_b128(loff + 16, gB + 8);

  v8f zz = {};
  v8f acc[4][2];
#pragma unroll
  for (int mi = 0; mi < 4; ++mi)
#pragma unroll
    for (int ni = 0; ni < 2; ++ni) acc[mi][ni] = zz;

  for (int kb = 0; kb < K; kb += 32) {
    const int cb = (kb >> 5) & 1;
    const int nb = kb + 32;
    if (nb < K) {
      const unsigned nl = loff + (unsigned)(((nb >> 5) & 1) * (128 * 32 * 2));
      async_b128(nl,      gB + nb);
      async_b128(nl + 16, gB + nb + 8);
      asm volatile("s_wait_asynccnt 0x2" ::: "memory");  // stage kb landed
    } else {
      asm volatile("s_wait_asynccnt 0x0" ::: "memory");
    }
    __syncthreads();

    v16bf bfr[2];
#pragma unroll
    for (int ni = 0; ni < 2; ++ni)
      bfr[ni] = load_b_frag(&lbs[cb][(n0w + ni * 16 + ln) * 32], half);
#pragma unroll
    for (int mi = 0; mi < 4; ++mi) {
      const bf16_t* ap = A + (size_t)(m0 + mi * 16 + ln) * lda + kb;
      v16bf afr = load_a_frag(ap, half);
      __builtin_prefetch(ap + 32, 0, 1);  // global_prefetch next K slice
      acc[mi][0] = wmma_bf(afr, bfr[0], acc[mi][0]);
      acc[mi][1] = wmma_bf(afr, bfr[1], acc[mi][1]);
    }
    __syncthreads();  // protect buffer reuse at kb+64
  }

#pragma unroll
  for (int mi = 0; mi < 4; ++mi) {
#pragma unroll
    for (int ni = 0; ni < 2; ++ni) {
      const int gcol = n0 + ni * 16 + ln;
      const float bv = (MODE == 3 || MODE == 4) ? bias[gcol] : 0.0f;
#pragma unroll
      for (int r = 0; r < 8; ++r) {
        const int grow = m0 + mi * 16 + half * 8 + r;  // C layout: m=r+8*half
        const float v = acc[mi][ni][r];
        if (MODE == 0) {
          outH[(size_t)grow * N + gcol] = (bf16_t)v;
        } else if (MODE == 1) {
          outH[(size_t)gcol * M + grow] = (bf16_t)v;
        } else if (MODE == 2) {
          outF[(size_t)grow * N + gcol] = resid[(size_t)grow * N + gcol] + v;
        } else if (MODE == 3) {
          const float tt = v + bv;
          outH[(size_t)grow * N + gcol] = (bf16_t)(tt > 0.0f ? tt : 0.01f * tt);
        } else {
          outF[(size_t)grow * N + gcol] =
              resid[(size_t)grow * N + gcol] + v + bv;
        }
      }
    }
  }
}

// ---------------------------------------------------------------------------
// Flash attention. Grid: (SEQ/64, NB*NH); 128 threads = 4 waves.
// Each wave owns 16 query rows; streams keys in blocks of 32 with online
// softmax. The stabilizer is the running max over the whole 16x32 score tile
// (valid for every row; softmax invariant to stabilizer choice), so the max
// reduction is one 5-level full-wave butterfly and the rescale factor is a
// single scalar exp. P (16x32) is re-laid C->A through a private 1KB LDS
// tile; row sums come from a ones-matrix WMMA on the reread P fragment.
// Q,K: bf16 (b,s,h*64+hd) row-major. Vt: bf16 (h*64+hd, b*S+s), ld=MROWS.
// ---------------------------------------------------------------------------
__global__ __launch_bounds__(128) void attn_kernel(
    const bf16_t* __restrict__ Q, const bf16_t* __restrict__ Kc,
    const bf16_t* __restrict__ Vt, bf16_t* __restrict__ O)
{
  const int lane = threadIdx.x & 31;
  const int wave = threadIdx.x >> 5;
  const int half = lane >> 4;
  const int ln   = lane & 15;
  const int b = blockIdx.y >> 4;   // NH == 16
  const int h = blockIdx.y & 15;
  const int q0 = blockIdx.x * 64 + wave * 16;

  const bf16_t* Qb = Q  + (size_t)b * SEQ * DM + (size_t)h * HDIM;
  const bf16_t* Kb = Kc + (size_t)b * SEQ * DM + (size_t)h * HDIM;
  const bf16_t* Vb = Vt + (size_t)h * HDIM * MROWS + (size_t)b * SEQ;

  v16bf qf[2];
#pragma unroll
  for (int c = 0; c < 2; ++c)
    qf[c] = load_a_frag(Qb + (size_t)(q0 + ln) * DM + c * 32, half);

  const v16bf onesf = ones_frag();
  v8f zz = {};
  v8f oacc[4];
#pragma unroll
  for (int nt = 0; nt < 4; ++nt) oacc[nt] = zz;
  float mr = -3.0e38f;   // running block-wide stabilizer (scalar)
  float lrow[8];
#pragma unroll
  for (int r = 0; r < 8; ++r) lrow[r] = 0.f;

  __shared__ __align__(16) bf16_t pbuf[4 * 512];  // 16x32 bf16 per wave
  bf16_t* pb = pbuf + wave * 512;

  const float scale = 0.03125f;  // 1/sqrt(D) = 1/32

  for (int kb = 0; kb < SEQ; kb += 32) {
    // ---- scores S = Q * K^T for 32 keys (2 n-tiles x 2 k-chunks) ----
    v8f sc[2]; sc[0] = zz; sc[1] = zz;
#pragma unroll
    for (int nt = 0; nt < 2; ++nt) {
      const bf16_t* kp = Kb + (size_t)(kb + nt * 16 + ln) * DM;
#pragma unroll
      for (int c = 0; c < 2; ++c)
        sc[nt] = wmma_bf(qf[c], load_b_frag(kp + c * 32, half), sc[nt]);
    }
    // ---- block-wide max: in-register tree + 5-level full-wave butterfly ----
    float bm = -3.0e38f;
#pragma unroll
    for (int r = 0; r < 8; ++r) {
      sc[0][r] *= scale;
      sc[1][r] *= scale;
      bm = fmaxf(bm, fmaxf(sc[0][r], sc[1][r]));
    }
#pragma unroll
    for (int off = 16; off >= 1; off >>= 1)
      bm = fmaxf(bm, __shfl_xor(bm, off));
    const float nm   = fmaxf(mr, bm);
    const float corr = __expf(mr - nm);   // single scalar rescale factor
    mr = nm;
    // ---- exponentiate into the LDS P tile (C layout: m=r+8*half, n=ln) ----
#pragma unroll
    for (int r = 0; r < 8; ++r) {
      pb[(half * 8 + r) * 32 + ln]      = (bf16_t)__expf(sc[0][r] - nm);
      pb[(half * 8 + r) * 32 + 16 + ln] = (bf16_t)__expf(sc[1][r] - nm);
    }
#pragma unroll
    for (int nt = 0; nt < 4; ++nt)
#pragma unroll
      for (int r = 0; r < 8; ++r) oacc[nt][r] *= corr;
    __syncthreads();
    // ---- reread P as A-fragment; ones-WMMA gives row sums in C layout ----
    v16bf pf = load_a_frag(pb + ln * 32, half);
    v8f rsum = wmma_bf(pf, onesf, zz);
#pragma unroll
    for (int r = 0; r < 8; ++r) lrow[r] = lrow[r] * corr + rsum[r];
    // ---- O += P * V : V^T gives B-fragments ----
#pragma unroll
    for (int nt = 0; nt < 4; ++nt) {
      v16bf vf = load_b_frag(Vb + (size_t)(nt * 16 + ln) * MROWS + kb, half);
      oacc[nt] = wmma_bf(pf, vf, oacc[nt]);
    }
    __syncthreads();
  }

  bf16_t* Ob = O + (size_t)b * SEQ * DM + (size_t)h * HDIM;
#pragma unroll
  for (int r = 0; r < 8; ++r) {
    const float inv = 1.0f / lrow[r];
#pragma unroll
    for (int nt = 0; nt < 4; ++nt)
      Ob[(size_t)(q0 + half * 8 + r) * DM + nt * 16 + ln] =
          (bf16_t)(oacc[nt][r] * inv);
  }
}

// ---------------------------------------------------------------------------
extern "C" void kernel_launch(void* const* d_in, const int* in_sizes, int n_in,
                              void* d_out, int out_size, void* d_ws, size_t ws_size,
                              hipStream_t stream)
{
  (void)in_sizes; (void)n_in; (void)out_size; (void)ws_size;
  const float* x   = (const float*)d_in[0];
  const float* Wq  = (const float*)d_in[1];
  const float* Wk  = (const float*)d_in[2];
  const float* Wv  = (const float*)d_in[3];
  const float* Wo  = (const float*)d_in[4];
  const float* W1  = (const float*)d_in[5];
  const float* b1  = (const float*)d_in[6];
  const float* W2  = (const float*)d_in[7];
  const float* b2  = (const float*)d_in[8];
  const float* g1  = (const float*)d_in[9];
  const float* be1 = (const float*)d_in[10];
  const float* g2  = (const float*)d_in[11];
  const float* be2 = (const float*)d_in[12];

  char* ws = (char*)d_ws;
  const size_t MBy = 1ull << 20;
  float*  xp  = (float*) (ws +   0 * MBy);  // x + posembed   32 MB
  bf16_t* hh  = (bf16_t*)(ws +  32 * MBy);  // LN1 out        16 MB (reused: o)
  bf16_t* qq  = (bf16_t*)(ws +  48 * MBy);  // Q              16 MB (reused: h2)
  bf16_t* kkk = (bf16_t*)(ws +  64 * MBy);  // K              16 MB
  bf16_t* vt  = (bf16_t*)(ws +  80 * MBy);  // V^T            16 MB
  float*  x2  = (float*) (ws +  96 * MBy);  // post-attn res  32 MB
  bf16_t* ff  = (bf16_t*)(ws + 128 * MBy);  // FFN hidden     48 MB
  bf16_t* WqT = (bf16_t*)(ws + 176 * MBy);
  bf16_t* WkT = (bf16_t*)(ws + 178 * MBy);
  bf16_t* WvT = (bf16_t*)(ws + 180 * MBy);
  bf16_t* WoT = (bf16_t*)(ws + 182 * MBy);
  bf16_t* W1T = (bf16_t*)(ws + 184 * MBy);  // 6 MB
  bf16_t* W2T = (bf16_t*)(ws + 190 * MBy);  // 6 MB
  bf16_t* oo  = hh;   // attention output reuses hh
  bf16_t* h2  = qq;   // LN2 output reuses qq

  dim3 tb(256);
  // weights: fp32 -> bf16, transposed (Bt = N x K)
  wconv_kernel<<<dim3(DM / 32, DM / 32),  tb, 0, stream>>>(Wq, WqT, DM, DM);
  wconv_kernel<<<dim3(DM / 32, DM / 32),  tb, 0, stream>>>(Wk, WkT, DM, DM);
  wconv_kernel<<<dim3(DM / 32, DM / 32),  tb, 0, stream>>>(Wv, WvT, DM, DM);
  wconv_kernel<<<dim3(DM / 32, DM / 32),  tb, 0, stream>>>(Wo, WoT, DM, DM);
  wconv_kernel<<<dim3(DM / 32, DFF / 32), tb, 0, stream>>>(W1, W1T, DM, DFF);
  wconv_kernel<<<dim3(DFF / 32, DM / 32), tb, 0, stream>>>(W2, W2T, DFF, DM);

  // pos-embed + LN1
  ln_kernel<true><<<MROWS, tb, 0, stream>>>(x, g1, be1, xp, hh);

  // QKV projections
  gemm_kernel<0><<<dim3(DM / 128, MROWS / 128), tb, 0, stream>>>(
      hh, DM, WqT, DM, nullptr, nullptr, nullptr, qq, MROWS, DM, DM);
  gemm_kernel<0><<<dim3(DM / 128, MROWS / 128), tb, 0, stream>>>(
      hh, DM, WkT, DM, nullptr, nullptr, nullptr, kkk, MROWS, DM, DM);
  gemm_kernel<1><<<dim3(DM / 128, MROWS / 128), tb, 0, stream>>>(
      hh, DM, WvT, DM, nullptr, nullptr, nullptr, vt, MROWS, DM, DM);

  // attention
  attn_kernel<<<dim3(SEQ / 64, NB * NH), dim3(128), 0, stream>>>(qq, kkk, vt, oo);

  // output projection + residual
  gemm_kernel<2><<<dim3(DM / 128, MROWS / 128), tb, 0, stream>>>(
      oo, DM, WoT, DM, nullptr, xp, x2, nullptr, MROWS, DM, DM);

  // LN2
  ln_kernel<false><<<MROWS, tb, 0, stream>>>(x2, g2, be2, nullptr, h2);

  // FFN
  gemm_kernel<3><<<dim3(DFF / 128, MROWS / 128), tb, 0, stream>>>(
      h2, DM, W1T, DM, b1, nullptr, nullptr, ff, MROWS, DFF, DM);
  gemm_kernel<4><<<dim3(DM / 128, MROWS / 128), tb, 0, stream>>>(
      ff, DFF, W2T, DFF, b2, x2, (float*)d_out, nullptr, MROWS, DM, DFF);
}
